// MPNN_81080392614044
// MI455X (gfx1250) — compile-verified, matrix operand order checked by reading, not run on hardware
//
#include <hip/hip_runtime.h>

#define ZN   8192
#define KNB  24
#define DIM  256

#define GAS __attribute__((address_space(1)))
#define LAS __attribute__((address_space(3)))

typedef __attribute__((ext_vector_type(16))) __bf16 v16bf;
typedef __attribute__((ext_vector_type(8)))  __bf16 v8bf;
typedef __attribute__((ext_vector_type(8)))  float  v8f;
typedef __attribute__((ext_vector_type(4)))  float  v4f;

typedef LAS __bf16* lbf16p;
typedef LAS float*  lfltp;

__device__ __forceinline__ v8f wmma_bf16(v16bf a, v16bf b, v8f c) {
  // 8 args: (neg_a, A, neg_b, B, c_mod, C, reuse_a, reuse_b)
  return __builtin_amdgcn_wmma_f32_16x16x32_bf16(false, a, false, b, (short)0, c,
                                                 false, false);
}

// A fragment (16x32 bf16) from LDS, row-major pitch 256.
// ISA layout: lanes 0-15 -> M=lane, K={k..k+7, k+16..k+23}; lanes 16-31 -> K base +8.
__device__ __forceinline__ v16bf frag_a(const LAS __bf16* buf, int row0, int koff) {
  const int lane = threadIdx.x & 31;
  const LAS __bf16* p = buf + (row0 + (lane & 15)) * DIM + koff + ((lane >> 4) << 3);
  v8bf c0 = *(const LAS v8bf*)p;
  v8bf c1 = *(const LAS v8bf*)(p + 16);
  v16bf a;
#pragma unroll
  for (int e = 0; e < 8; ++e) { a[e] = c0[e]; a[8 + e] = c1[e]; }
  return a;
}

// B fragment (32x16 bf16) from transposed weights Wt[N][K] in global memory.
// 32-bit offset math -> global_load with SGPR base.
__device__ __forceinline__ v16bf frag_b(const GAS __bf16* __restrict__ wt, int pitch,
                                        int n0, int koff) {
  const int lane = threadIdx.x & 31;
  const int off = (n0 + (lane & 15)) * pitch + koff + ((lane >> 4) << 3);
  v8bf c0 = *(const GAS v8bf*)(wt + off);
  v8bf c1 = *(const GAS v8bf*)(wt + off + 16);
  v16bf b;
#pragma unroll
  for (int e = 0; e < 8; ++e) { b[e] = c0[e]; b[8 + e] = c1[e]; }
  return b;
}

__device__ __forceinline__ float silu_f(float v) { return v / (1.f + __expf(-v)); }

// 32x256 dense layer on a 256-thread block (8 wave32). in pitch 256 bf16.
// MODEF==0: bf16 out (+optional SiLU). MODEF==1: f32 out, no SiLU.
template <int MODEF>
__device__ __forceinline__ void dense32(const LAS __bf16* inb,
                                        const GAS __bf16* __restrict__ wt, int wpitch,
                                        int kdim, const GAS float* __restrict__ bias,
                                        lbf16p outB, lfltp outF, bool silu) {
  __syncthreads();  // previous layer's outputs visible
  const int wave = threadIdx.x >> 5, lane = threadIdx.x & 31;
#pragma unroll
  for (int i = 0; i < 2; ++i) {
    const int nt = wave * 2 + i;
    v8f c0 = {};
    v8f c1 = {};
    for (int k = 0; k < kdim; k += 32) {
      v16bf b  = frag_b(wt, wpitch, nt * 16, k);
      v16bf a0 = frag_a(inb, 0, k);
      v16bf a1 = frag_a(inb, 16, k);
      c0 = wmma_bf16(a0, b, c0);
      c1 = wmma_bf16(a1, b, c1);
    }
    const int n  = nt * 16 + (lane & 15);
    const int mr = (lane >> 4) << 3;
    const float bv = bias[n];
#pragma unroll
    for (int r = 0; r < 8; ++r) {
      float v0 = c0[r] + bv, v1 = c1[r] + bv;
      if (MODEF == 0) {
        if (silu) { v0 = silu_f(v0); v1 = silu_f(v1); }
        outB[(mr + r) * DIM + n]      = (__bf16)v0;
        outB[(mr + r + 16) * DIM + n] = (__bf16)v1;
      } else {
        outF[(mr + r) * DIM + n]      = v0;
        outF[(mr + r + 16) * DIM + n] = v1;
      }
    }
  }
}

// Layers 1-3 of the 768->256->256->256 message MLP for node i.
// Layer 1 processes the 768-wide input as 3 segments of 256 staged in bufS.
// On return, layer-3 activations (bf16) sit in bufA.
__device__ __forceinline__ void msg_mlp_front(
    const GAS float* __restrict__ nodesSrc, const GAS float* __restrict__ edges,
    const GAS long long* __restrict__ nbrs, int i,
    const GAS __bf16* __restrict__ w0t, const GAS float* __restrict__ b0,
    const GAS __bf16* __restrict__ w1t, const GAS float* __restrict__ b1,
    const GAS __bf16* __restrict__ w2t, const GAS float* __restrict__ b2,
    lbf16p bufA, lbf16p bufS, lbf16p bufB) {
  const int wave = threadIdx.x >> 5, lane = threadIdx.x & 31;
  const int t  = threadIdx.x;
  const int m  = t >> 3;                 // staging row 0..31
  const int mm = m < KNB ? m : KNB - 1;  // clamp padding rows
  const int c0 = (t & 7) * 32;           // staging col base

  v8f acc[2][2] = {};
  for (int seg = 0; seg < 3; ++seg) {
    __syncthreads();  // all reads of previous segment done
    const GAS float* rsrc;
    if (seg == 0)
      rsrc = nodesSrc + (size_t)i * DIM;  // x_i broadcast to all rows
    else if (seg == 1)
      rsrc = nodesSrc + (size_t)nbrs[(size_t)i * KNB + mm] * DIM;  // x_j gather
    else
      rsrc = edges + ((size_t)i * KNB + mm) * DIM;  // e_ij
    const GAS v4f* s4 = (const GAS v4f*)(rsrc + c0);
    lbf16p d = bufS + m * DIM + c0;
#pragma unroll
    for (int q = 0; q < 8; ++q) {
      v4f v = s4[q];
      d[q * 4 + 0] = (__bf16)v[0];
      d[q * 4 + 1] = (__bf16)v[1];
      d[q * 4 + 2] = (__bf16)v[2];
      d[q * 4 + 3] = (__bf16)v[3];
    }
    __syncthreads();
#pragma unroll
    for (int ii = 0; ii < 2; ++ii) {
      const int nt = wave * 2 + ii;
      for (int k = 0; k < DIM; k += 32) {
        v16bf b  = frag_b(w0t, 3 * DIM, nt * 16, seg * DIM + k);
        v16bf a0 = frag_a(bufS, 0, k);
        v16bf a1 = frag_a(bufS, 16, k);
        acc[ii][0] = wmma_bf16(a0, b, acc[ii][0]);
        acc[ii][1] = wmma_bf16(a1, b, acc[ii][1]);
      }
    }
  }
  // Layer-1 epilogue: bias + SiLU -> bufA
#pragma unroll
  for (int ii = 0; ii < 2; ++ii) {
    const int nt = wave * 2 + ii;
    const int n  = nt * 16 + (lane & 15);
    const int mr = (lane >> 4) << 3;
    const float bv = b0[n];
#pragma unroll
    for (int r = 0; r < 8; ++r) {
      bufA[(mr + r) * DIM + n]      = (__bf16)silu_f(acc[ii][0][r] + bv);
      bufA[(mr + r + 16) * DIM + n] = (__bf16)silu_f(acc[ii][1][r] + bv);
    }
  }
  dense32<0>(bufA, w1t, DIM, DIM, b1, bufB, nullptr, true);  // L2: bufA->bufB
  dense32<0>(bufB, w2t, DIM, DIM, b2, bufA, nullptr, true);  // L3: bufB->bufA
}

// ---------------- Kernel 1: weight transpose + bf16 convert ----------------
__global__ void k_wt(const float* __restrict__ src_, __bf16* __restrict__ dst_,
                     int K, int N) {
  const GAS float* src = (const GAS float*)src_;
  GAS __bf16* dst = (GAS __bf16*)dst_;
  const int idx = blockIdx.x * 256 + threadIdx.x;  // over N*K (dst layout [n][k])
  if (idx >= K * N) return;
  const int n = idx / K, k = idx - n * K;
  dst[idx] = (__bf16)src[(size_t)k * N + n];
}

// ---------- Kernel 2: per-node message MLP + masked sum + LN1 ----------
__global__ void __launch_bounds__(256) k_node_msg(
    const float* __restrict__ nodes_, const float* __restrict__ edges_,
    const long long* __restrict__ nbrs_, const float* __restrict__ nbr_mask_,
    const __bf16* __restrict__ w0t_, const float* __restrict__ b0_,
    const __bf16* __restrict__ w1t_, const float* __restrict__ b1_,
    const __bf16* __restrict__ w2t_, const float* __restrict__ b2_,
    const __bf16* __restrict__ w3t_, const float* __restrict__ b3_,
    const float* __restrict__ g1_, const float* __restrict__ be1_,
    float* __restrict__ nodes_ln1_) {
  __shared__ __align__(16) unsigned char smem_[49152];
  __shared__ float mask_s[32];
  __shared__ float rsum[8], rsq[8];
  lbf16p bufA = (lbf16p)(smem_);            // 16KB
  lbf16p bufS = (lbf16p)(smem_ + 16384);    // 16KB
  lbf16p bufB = (lbf16p)(smem_ + 32768);    // 16KB
  lfltp  outF = (lfltp)(smem_ + 16384);     // 32KB overlay of bufS+bufB

  const GAS float* nodes = (const GAS float*)nodes_;
  const GAS float* edges = (const GAS float*)edges_;
  const GAS long long* nbrs = (const GAS long long*)nbrs_;
  const GAS float* nbr_mask = (const GAS float*)nbr_mask_;
  const GAS float* g1 = (const GAS float*)g1_;
  const GAS float* be1 = (const GAS float*)be1_;
  GAS float* nodes_ln1 = (GAS float*)nodes_ln1_;

  const int i = blockIdx.x;
  const int t = threadIdx.x;
  if (t < 32) mask_s[t] = (t < KNB) ? nbr_mask[(size_t)i * KNB + t] : 0.f;

  msg_mlp_front(nodes, edges, nbrs, i, (const GAS __bf16*)w0t_, (const GAS float*)b0_,
                (const GAS __bf16*)w1t_, (const GAS float*)b1_,
                (const GAS __bf16*)w2t_, (const GAS float*)b2_, bufA, bufS, bufB);
  dense32<1>(bufA, (const GAS __bf16*)w3t_, DIM, DIM, (const GAS float*)b3_, nullptr,
             outF, false);  // L4 -> f32
  __syncthreads();

  // deterministic masked sum over 24 rows + residual + LayerNorm(256)
  const int wave = t >> 5, lane = t & 31;
  float y = nodes[(size_t)i * DIM + t];
#pragma unroll 4
  for (int m = 0; m < KNB; ++m) y += outF[m * DIM + t] * mask_s[m];
  float s = y, q = y * y;
#pragma unroll
  for (int off = 16; off; off >>= 1) {
    s += __shfl_xor(s, off, 32);
    q += __shfl_xor(q, off, 32);
  }
  if (lane == 0) { rsum[wave] = s; rsq[wave] = q; }
  __syncthreads();
  s = 0.f; q = 0.f;
#pragma unroll
  for (int w = 0; w < 8; ++w) { s += rsum[w]; q += rsq[w]; }
  const float mu   = s * (1.f / 256.f);
  const float var  = q * (1.f / 256.f) - mu * mu;
  const float rstd = rsqrtf(var + 1e-5f);
  nodes_ln1[(size_t)i * DIM + t] = (y - mu) * rstd * g1[t] + be1[t];
}

// ---------- Kernel 3: FFN (256->1024->256) + residual + LN2 ----------
__global__ void __launch_bounds__(256) k_ffn(
    const float* __restrict__ xin_, const __bf16* __restrict__ fw0t_,
    const float* __restrict__ fb0_, const __bf16* __restrict__ fw1t_,
    const float* __restrict__ fb1_, const float* __restrict__ g2_,
    const float* __restrict__ b2_, float* __restrict__ nodes_out_) {
  __shared__ __align__(16) unsigned char smem_[32768];
  lbf16p bufA = (lbf16p)(smem_);            // 16KB  (x, bf16)
  lbf16p bufH = (lbf16p)(smem_ + 16384);    // 16KB  (H chunk, bf16)
  lfltp  bufO = (lfltp)(smem_);             // 32KB overlay (final f32)

  const GAS float* xin = (const GAS float*)xin_;
  const GAS __bf16* fw0t = (const GAS __bf16*)fw0t_;
  const GAS float* fb0 = (const GAS float*)fb0_;
  const GAS __bf16* fw1t = (const GAS __bf16*)fw1t_;
  const GAS float* fb1 = (const GAS float*)fb1_;
  const GAS float* g2 = (const GAS float*)g2_;
  const GAS float* b2 = (const GAS float*)b2_;
  GAS float* nodes_out = (GAS float*)nodes_out_;

  const int base = blockIdx.x * 32;
  const int t = threadIdx.x, wave = t >> 5, lane = t & 31;

  {  // stage x rows -> bufA
    const int m = t >> 3, c0 = (t & 7) * 32;
    const GAS v4f* s4 = (const GAS v4f*)(xin + (size_t)(base + m) * DIM + c0);
    lbf16p d = bufA + m * DIM + c0;
#pragma unroll
    for (int q = 0; q < 8; ++q) {
      v4f v = s4[q];
      d[q * 4 + 0] = (__bf16)v[0];
      d[q * 4 + 1] = (__bf16)v[1];
      d[q * 4 + 2] = (__bf16)v[2];
      d[q * 4 + 3] = (__bf16)v[3];
    }
  }

  v8f c2[2][2] = {};
  for (int ch = 0; ch < 4; ++ch) {
    // GEMM1 chunk: H[:, ch*256..) = silu(x @ W0t_chunk)  (dense32 syncs on entry)
    dense32<0>(bufA, fw0t + ch * 256 * DIM, DIM, DIM, fb0 + ch * 256, bufH, nullptr,
               true);
    __syncthreads();
    // GEMM2 partial accumulation over this chunk's K range
#pragma unroll
    for (int ii = 0; ii < 2; ++ii) {
      const int nt = wave * 2 + ii;
      for (int k = 0; k < DIM; k += 32) {
        v16bf b  = frag_b(fw1t, 4 * DIM, nt * 16, ch * DIM + k);
        v16bf a0 = frag_a(bufH, 0, k);
        v16bf a1 = frag_a(bufH, 16, k);
        c2[ii][0] = wmma_bf16(a0, b, c2[ii][0]);
        c2[ii][1] = wmma_bf16(a1, b, c2[ii][1]);
      }
    }
  }
  __syncthreads();
  // epilogue: bias + residual(x) -> bufO (f32)
#pragma unroll
  for (int ii = 0; ii < 2; ++ii) {
    const int nt = wave * 2 + ii;
    const int n  = nt * 16 + (lane & 15);
    const int mr = (lane >> 4) << 3;
    const float bv = fb1[n];
#pragma unroll
    for (int r = 0; r < 8; ++r) {
      const int m0 = mr + r, m1 = mr + r + 16;
      bufO[m0 * DIM + n] = c2[ii][0][r] + bv + xin[(size_t)(base + m0) * DIM + n];
      bufO[m1 * DIM + n] = c2[ii][1][r] + bv + xin[(size_t)(base + m1) * DIM + n];
    }
  }
  __syncthreads();
  // row-wise LayerNorm
#pragma unroll
  for (int j = 0; j < 4; ++j) {
    const int m = wave * 4 + j;
    float v[8], s = 0.f, q = 0.f;
#pragma unroll
    for (int e = 0; e < 8; ++e) {
      v[e] = bufO[m * DIM + lane * 8 + e];
      s += v[e];
      q += v[e] * v[e];
    }
#pragma unroll
    for (int off = 16; off; off >>= 1) {
      s += __shfl_xor(s, off, 32);
      q += __shfl_xor(q, off, 32);
    }
    const float mu   = s * (1.f / 256.f);
    const float rstd = rsqrtf(q * (1.f / 256.f) - mu * mu + 1e-5f);
#pragma unroll
    for (int e = 0; e < 8; ++e) {
      const int n = lane * 8 + e;
      nodes_out[(size_t)(base + m) * DIM + n] = (v[e] - mu) * rstd * g2[n] + b2[n];
    }
  }
}

// ---------- Kernel 4: per-node edge MLP + residual + edge LN ----------
__global__ void __launch_bounds__(256) k_edge_msg(
    const float* __restrict__ nodes_fin_, const float* __restrict__ edges_,
    const long long* __restrict__ nbrs_,
    const __bf16* __restrict__ w0t_, const float* __restrict__ b0_,
    const __bf16* __restrict__ w1t_, const float* __restrict__ b1_,
    const __bf16* __restrict__ w2t_, const float* __restrict__ b2_,
    const __bf16* __restrict__ w3t_, const float* __restrict__ b3_,
    const float* __restrict__ ge_, const float* __restrict__ bee_,
    float* __restrict__ edges_out_) {
  __shared__ __align__(16) unsigned char smem_[49152];
  lbf16p bufA = (lbf16p)(smem_);
  lbf16p bufS = (lbf16p)(smem_ + 16384);
  lbf16p bufB = (lbf16p)(smem_ + 32768);
  lfltp  outF = (lfltp)(smem_ + 16384);

  const GAS float* nodes_fin = (const GAS float*)nodes_fin_;
  const GAS float* edges = (const GAS float*)edges_;
  const GAS long long* nbrs = (const GAS long long*)nbrs_;
  const GAS float* ge = (const GAS float*)ge_;
  const GAS float* bee = (const GAS float*)bee_;
  GAS float* edges_out = (GAS float*)edges_out_;

  const int i = blockIdx.x;

  msg_mlp_front(nodes_fin, edges, nbrs, i, (const GAS __bf16*)w0t_,
                (const GAS float*)b0_, (const GAS __bf16*)w1t_,
                (const GAS float*)b1_, (const GAS __bf16*)w2t_,
                (const GAS float*)b2_, bufA, bufS, bufB);
  dense32<1>(bufA, (const GAS __bf16*)w3t_, DIM, DIM, (const GAS float*)b3_, nullptr,
             outF, false);
  __syncthreads();

  const int wave = threadIdx.x >> 5, lane = threadIdx.x & 31;
#pragma unroll
  for (int j = 0; j < 4; ++j) {
    const int m = wave * 4 + j;
    if (m < KNB) {
      float v[8], s = 0.f, q = 0.f;
#pragma unroll
      for (int e = 0; e < 8; ++e) {
        const int n = lane * 8 + e;
        v[e] = edges[((size_t)i * KNB + m) * DIM + n] + outF[m * DIM + n];
        s += v[e];
        q += v[e] * v[e];
      }
#pragma unroll
      for (int off = 16; off; off >>= 1) {
        s += __shfl_xor(s, off, 32);
        q += __shfl_xor(q, off, 32);
      }
      const float mu   = s * (1.f / 256.f);
      const float rstd = rsqrtf(q * (1.f / 256.f) - mu * mu + 1e-5f);
#pragma unroll
      for (int e = 0; e < 8; ++e) {
        const int n = lane * 8 + e;
        edges_out[((size_t)i * KNB + m) * DIM + n] = (v[e] - mu) * rstd * ge[n] + bee[n];
      }
    }
  }
}

// --------------------------- host launcher ---------------------------
extern "C" void kernel_launch(void* const* d_in, const int* in_sizes, int n_in,
                              void* d_out, int out_size, void* d_ws, size_t ws_size,
                              hipStream_t stream) {
  (void)in_sizes; (void)n_in; (void)out_size; (void)ws_size;
  const float*     nodes    = (const float*)d_in[0];
  const float*     edges    = (const float*)d_in[1];
  const long long* nbrs     = (const long long*)d_in[2];  // jnp.int64
  const float*     nbr_mask = (const float*)d_in[3];
  const float* nW[4] = {(const float*)d_in[4], (const float*)d_in[6],
                        (const float*)d_in[8], (const float*)d_in[10]};
  const float* nb[4] = {(const float*)d_in[5], (const float*)d_in[7],
                        (const float*)d_in[9], (const float*)d_in[11]};
  const float* fW0 = (const float*)d_in[12];
  const float* fb0 = (const float*)d_in[13];
  const float* fW1 = (const float*)d_in[14];
  const float* fb1 = (const float*)d_in[15];
  const float* eW[4] = {(const float*)d_in[16], (const float*)d_in[18],
                        (const float*)d_in[20], (const float*)d_in[22]};
  const float* eb[4] = {(const float*)d_in[17], (const float*)d_in[19],
                        (const float*)d_in[21], (const float*)d_in[23]};
  const float* g1  = (const float*)d_in[24];
  const float* be1 = (const float*)d_in[25];
  const float* g2  = (const float*)d_in[26];
  const float* b2  = (const float*)d_in[27];
  const float* ge  = (const float*)d_in[28];
  const float* bee = (const float*)d_in[29];

  unsigned char* ws = (unsigned char*)d_ws;
  size_t off = 0;
  auto carve = [&](size_t bytes) -> void* {
    void* p = ws + off;
    off += (bytes + 255) & ~(size_t)255;
    return p;
  };
  __bf16* nW0t = (__bf16*)carve((size_t)768 * 256 * 2);
  __bf16* nW1t = (__bf16*)carve((size_t)256 * 256 * 2);
  __bf16* nW2t = (__bf16*)carve((size_t)256 * 256 * 2);
  __bf16* nW3t = (__bf16*)carve((size_t)256 * 256 * 2);
  __bf16* fW0t = (__bf16*)carve((size_t)1024 * 256 * 2);
  __bf16* fW1t = (__bf16*)carve((size_t)256 * 1024 * 2);
  __bf16* eW0t = (__bf16*)carve((size_t)768 * 256 * 2);
  __bf16* eW1t = (__bf16*)carve((size_t)256 * 256 * 2);
  __bf16* eW2t = (__bf16*)carve((size_t)256 * 256 * 2);
  __bf16* eW3t = (__bf16*)carve((size_t)256 * 256 * 2);
  float* nodes_ln1 = (float*)carve((size_t)ZN * DIM * 4);

  auto xpose = [&](const float* src, __bf16* dst, int K, int N) {
    const int tot = K * N;
    k_wt<<<(tot + 255) / 256, 256, 0, stream>>>(src, dst, K, N);
  };
  xpose(nW[0], nW0t, 768, 256);
  xpose(nW[1], nW1t, 256, 256);
  xpose(nW[2], nW2t, 256, 256);
  xpose(nW[3], nW3t, 256, 256);
  xpose(fW0, fW0t, 256, 1024);
  xpose(fW1, fW1t, 1024, 256);
  xpose(eW[0], eW0t, 768, 256);
  xpose(eW[1], eW1t, 256, 256);
  xpose(eW[2], eW2t, 256, 256);
  xpose(eW[3], eW3t, 256, 256);

  float* nodes_out = (float*)d_out;
  float* edges_out = nodes_out + (size_t)ZN * DIM;

  k_node_msg<<<ZN, 256, 0, stream>>>(nodes, edges, nbrs, nbr_mask, nW0t, nb[0],
                                     nW1t, nb[1], nW2t, nb[2], nW3t, nb[3], g1,
                                     be1, nodes_ln1);
  k_ffn<<<ZN / 32, 256, 0, stream>>>(nodes_ln1, fW0t, fb0, fW1t, fb1, g2, b2,
                                     nodes_out);
  k_edge_msg<<<ZN, 256, 0, stream>>>(nodes_out, edges, nbrs, eW0t, eb[0], eW1t,
                                     eb[1], eW2t, eb[2], eW3t, eb[3], ge, bee,
                                     edges_out);
}
